// HPNCritic_1546188226866
// MI455X (gfx1250) — matrix operand order repaired and checked by reading, not compiled, not run on hardware
//
#include <hip/hip_runtime.h>
#include <hip/hip_bf16.h>

// ---------------------------------------------------------------------------
// HPN critic, restructured:  merged = Pcat @ WmT^T  (one bf16 WMMA GEMM)
//   Pcat[b, kd] = sum_n mask * hidext[b,n,k] * fext[b,n,d]    (kd = k*ND+d)
//   WmT[j, kd]  = sum_head softmax(merger_w)[head,j] * W2ext[k, d*1024+head*256+j]
// followed by relu(+self_out), WMMA GEMM vs W_o1^T, relu(+b_o1), GEMV W_o2.
// ---------------------------------------------------------------------------

typedef __attribute__((ext_vector_type(16))) __bf16 v16bf;
typedef __attribute__((ext_vector_type(8)))  float  v8f;

#define BATCH 1024
#define CH    256            // HID in reference
#define KA    1419           // 129 * 11  (ally: 128 hid + 1 bias) x (10 feat + 1)
#define KA_P  1440           // padded to multiple of 32
#define KV    903            // 129 * 7   (adv: 128 hid + 1 bias) x (6 feat + 1)
#define KV_P  928
#define KC    (KA_P + KV_P)  // 2368 = 74 * 32

// workspace layout (bytes); hipMalloc base is >=256B aligned
constexpr size_t SZ_WMT   = (size_t)CH * KC * 2;        // bf16 (256 x 2368)
constexpr size_t SZ_WO1T  = (size_t)CH * CH * 2;        // bf16 (256 x 256)
constexpr size_t SZ_PCAT  = (size_t)BATCH * KC * 2;     // bf16 (1024 x 2368)
constexpr size_t SZ_SELF  = (size_t)BATCH * CH * 4;     // f32
constexpr size_t SZ_X     = (size_t)BATCH * CH * 2;     // bf16
constexpr size_t OFF_WMT  = 0;
constexpr size_t OFF_WO1T = OFF_WMT + SZ_WMT;
constexpr size_t OFF_PCAT = OFF_WO1T + SZ_WO1T;
constexpr size_t OFF_SELF = OFF_PCAT + SZ_PCAT;
constexpr size_t OFF_X    = OFF_SELF + SZ_SELF;
constexpr size_t OFF_Y    = OFF_X + SZ_X;

// --------------------------------------------------------------------------
// Fold softmax(merger_w) over heads into the hyper weights -> WmT (bf16).
// grid = KC blocks (kd), block = 256 threads (j).
// --------------------------------------------------------------------------
__global__ void fold_wmt(const float* __restrict__ W_ha2, const float* __restrict__ b_ha2,
                         const float* __restrict__ W_hv2, const float* __restrict__ b_hv2,
                         const float* __restrict__ merger_w, __bf16* __restrict__ WmT)
{
    int kd = blockIdx.x;
    int j  = threadIdx.x;
    float m0 = merger_w[0 * CH + j], m1 = merger_w[1 * CH + j];
    float m2 = merger_w[2 * CH + j], m3 = merger_w[3 * CH + j];
    float mx = fmaxf(fmaxf(m0, m1), fmaxf(m2, m3));
    float e0 = __expf(m0 - mx), e1 = __expf(m1 - mx);
    float e2 = __expf(m2 - mx), e3 = __expf(m3 - mx);
    float inv = 1.0f / (e0 + e1 + e2 + e3);
    float sm[4] = { e0 * inv, e1 * inv, e2 * inv, e3 * inv };

    float v = 0.0f;
    if (kd < KA_P) {
        if (kd < KA) {
            int k = kd / 11, d = kd % 11;
            const float* row = (k < 128) ? (W_ha2 + (size_t)k * 11264) : b_ha2;
            #pragma unroll
            for (int h = 0; h < 4; ++h) v += sm[h] * row[d * 1024 + h * CH + j];
        }
    } else {
        int r = kd - KA_P;
        if (r < KV) {
            int k = r / 7, d = r % 7;
            const float* row = (k < 128) ? (W_hv2 + (size_t)k * 7168) : b_hv2;
            #pragma unroll
            for (int h = 0; h < 4; ++h) v += sm[h] * row[d * 1024 + h * CH + j];
        }
    }
    WmT[(size_t)j * KC + kd] = (__bf16)v;
}

// W_o1 (256x256 f32) -> transposed bf16.  grid=256 (j), block=256 (k)
__global__ void transpose_o1(const float* __restrict__ W_o1, __bf16* __restrict__ Wo1T)
{
    int j = blockIdx.x, k = threadIdx.x;
    Wo1T[(size_t)j * CH + k] = (__bf16)W_o1[(size_t)k * CH + j];
}

// --------------------------------------------------------------------------
// Per-batch-row features: hypernet hidden layers (leaky relu) and masked
// outer-product features Pcat (bf16), plus self branch (f32).
// grid = BATCH, block = 128 (thread = hidden unit k).
// --------------------------------------------------------------------------
__global__ void feat_kernel(const float* __restrict__ state, const float* __restrict__ action,
                            const float* __restrict__ W_self, const float* __restrict__ b_self,
                            const float* __restrict__ W_ha1,  const float* __restrict__ b_ha1,
                            const float* __restrict__ W_hv1,  const float* __restrict__ b_hv1,
                            const int* __restrict__ agent_id_p,
                            __bf16* __restrict__ Pcat, float* __restrict__ self_out)
{
    int b = blockIdx.x, tid = threadIdx.x;
    __shared__ float st[131];
    __shared__ float act[50];
    for (int i = tid; i < 131; i += 128) st[i]  = state[(size_t)b * 131 + i];
    for (int i = tid; i < 50;  i += 128) act[i] = action[(size_t)b * 50 + i];
    __syncthreads();
    const int aid = agent_id_p[0];   // little-endian: low 32 bits valid for i32/i64

    // masks, feature accessors
    float m01[9];
    #pragma unroll
    for (int n = 0; n < 9; ++n) m01[n] = (st[112 + n] < 0.5f) ? 0.0f : 1.0f;
    auto af = [&](int n, int d) -> float {
        int oidx = (n < aid) ? n : n + 1;
        return (d < 5) ? st[7 + n * 5 + d] : act[oidx * 5 + (d - 5)];
    };
    auto vf = [&](int n, int d) -> float { return st[52 + n * 6 + d]; };

    const size_t base = (size_t)b * KC;

    // ---- ally branch: hid_a[n] for hidden unit k=tid, then P entries -----
    {
        float hidA[9];
        #pragma unroll
        for (int n = 0; n < 9; ++n) {
            float h = b_ha1[tid];
            #pragma unroll
            for (int d = 0; d < 10; ++d) h += af(n, d) * W_ha1[d * 128 + tid];
            hidA[n] = (h > 0.0f) ? h : 0.01f * h;      // leaky_relu(0.01)
        }
        #pragma unroll
        for (int d = 0; d < 11; ++d) {
            float s = 0.0f;
            #pragma unroll
            for (int n = 0; n < 9; ++n) s += m01[n] * hidA[n] * ((d < 10) ? af(n, d) : 1.0f);
            Pcat[base + tid * 11 + d] = (__bf16)s;
        }
        if (tid < 11) {                    // hid-bias row (k=128): cols 1408..1418
            float s = 0.0f;
            #pragma unroll
            for (int n = 0; n < 9; ++n) s += m01[n] * ((tid < 10) ? af(n, tid) : 1.0f);
            Pcat[base + 1408 + tid] = (__bf16)s;
        } else if (tid < 32) {             // pad cols 1419..1439
            Pcat[base + 1408 + tid] = (__bf16)0.0f;
        }
    }

    // ---- adversary branch ------------------------------------------------
    {
        float hidV[10];
        #pragma unroll
        for (int n = 0; n < 10; ++n) {
            float h = b_hv1[tid];
            #pragma unroll
            for (int d = 0; d < 6; ++d) h += vf(n, d) * W_hv1[d * 128 + tid];
            hidV[n] = (h > 0.0f) ? h : 0.01f * h;
        }
        #pragma unroll
        for (int d = 0; d < 7; ++d) {
            float s = 0.0f;
            #pragma unroll
            for (int n = 0; n < 10; ++n) s += hidV[n] * ((d < 6) ? vf(n, d) : 1.0f);
            Pcat[base + KA_P + tid * 7 + d] = (__bf16)s;
        }
        if (tid < 7) {                     // bias row: cols 896..902 of adv part
            float s = 0.0f;
            #pragma unroll
            for (int n = 0; n < 10; ++n) s += (tid < 6) ? vf(n, tid) : 1.0f;
            Pcat[base + KA_P + 896 + tid] = (__bf16)s;
        } else if (tid < 32) {             // pad cols 903..927
            Pcat[base + KA_P + 896 + tid] = (__bf16)0.0f;
        }
    }

    // ---- self branch: self_feats(12) @ W_self + b_self -------------------
    #pragma unroll
    for (int jj = 0; jj < 2; ++jj) {
        int j = tid + jj * 128;
        float s = b_self[j];
        #pragma unroll
        for (int d = 0; d < 12; ++d) {
            float f = (d < 7) ? st[d] : act[aid * 5 + (d - 7)];
            s += f * W_self[d * CH + j];
        }
        self_out[(size_t)b * CH + j] = s;
    }
}

// --------------------------------------------------------------------------
// bf16 WMMA GEMM: C(bf16, MxCH) = relu( A(M x K) @ Bm^T(CH x K) [+addf][+bias] )
// A row-major, Bm stored transposed (per-column K-contiguous -> packed loads).
// grid = (M/16, CH/64), block = 128 (4 waves, one 16x16 tile each).
// Fragments (wave32): A/B use contiguous k = 16*hi + e, same for both; C/D layout per CDNA5 ISA 7.12.2.
// --------------------------------------------------------------------------
__global__ void wmma_gemm(const __bf16* __restrict__ A, const __bf16* __restrict__ Bm,
                          int K,
                          const float* __restrict__ addf,   // MxCH f32 or nullptr
                          const float* __restrict__ bias,   // CH f32 or nullptr
                          __bf16* __restrict__ C)
{
    const int lane = threadIdx.x & 31;
    const int wave = threadIdx.x >> 5;
    const int rowBase = blockIdx.x * 16;
    const int colBase = (blockIdx.y * 4 + wave) * 16;
    const int m  = lane & 15;
    const int hi = lane >> 4;

    const __bf16* Ap = A  + (size_t)(rowBase + m) * K;
    const __bf16* Bp = Bm + (size_t)(colBase + m) * K;   // lane's N-column, K-contig

    v8f acc = {};
    for (int kt = 0; kt < K; kt += 32) {
        v16bf a, bb;
        #pragma unroll
        for (int e = 0; e < 16; ++e) {                   // A 16x32: lanes0-15 K0-15, 16-31 K16-31
            int r = e >> 1, c = e & 1;
            int kA = hi * 16 + 2 * r + c;
            a[e] = Ap[kt + kA];
        }
        #pragma unroll
        for (int e = 0; e < 16; ++e) {                   // B 32x16: lanes0-15 K0-15, 16-31 K16-31
            int r = e >> 1, c = e & 1;
            int kB = hi * 16 + 2 * r + c;
            bb[e] = Bp[kt + kB];
        }
        acc = __builtin_amdgcn_wmma_f32_16x16x32_bf16(false, a, false, bb,
                                                      (short)0, acc, false, false);
    }

    // C/D layout: VGPR r -> M = r + 8*hi, N = lane&15
    #pragma unroll
    for (int r = 0; r < 8; ++r) {
        int row = rowBase + r + 8 * hi;
        int j   = colBase + (lane & 15);
        float v = acc[r];
        if (addf) v += addf[(size_t)row * CH + j];
        if (bias) v += bias[j];
        v = fmaxf(v, 0.0f);
        C[(size_t)row * CH + j] = (__bf16)v;
    }
}

// Final projection: out[b] = Y[b,:] . W_o2 + b_o2.  One wave32 per row.
__global__ void out_gemv(const __bf16* __restrict__ Y, const float* __restrict__ W_o2,
                         const float* __restrict__ b_o2, float* __restrict__ out)
{
    int lane = threadIdx.x & 31;
    int row  = blockIdx.x * 4 + (threadIdx.x >> 5);
    float p = 0.0f;
    #pragma unroll
    for (int i = 0; i < 8; ++i) {
        int j = lane + 32 * i;
        p += (float)Y[(size_t)row * CH + j] * W_o2[j];
    }
    #pragma unroll
    for (int off = 16; off > 0; off >>= 1) p += __shfl_xor(p, off, 32);
    if (lane == 0) out[row] = p + b_o2[0];
}

// --------------------------------------------------------------------------
extern "C" void kernel_launch(void* const* d_in, const int* in_sizes, int n_in,
                              void* d_out, int out_size, void* d_ws, size_t ws_size,
                              hipStream_t stream)
{
    (void)in_sizes; (void)n_in; (void)out_size; (void)ws_size;
    const float* state    = (const float*)d_in[0];
    const float* action   = (const float*)d_in[1];
    const float* W_self   = (const float*)d_in[2];
    const float* b_self   = (const float*)d_in[3];
    const float* W_ha1    = (const float*)d_in[4];
    const float* b_ha1    = (const float*)d_in[5];
    const float* W_ha2    = (const float*)d_in[6];
    const float* b_ha2    = (const float*)d_in[7];
    const float* W_hv1    = (const float*)d_in[8];
    const float* b_hv1    = (const float*)d_in[9];
    const float* W_hv2    = (const float*)d_in[10];
    const float* b_hv2    = (const float*)d_in[11];
    const float* merger_w = (const float*)d_in[12];
    const float* W_o1     = (const float*)d_in[13];
    const float* b_o1     = (const float*)d_in[14];
    const float* W_o2     = (const float*)d_in[15];
    const float* b_o2     = (const float*)d_in[16];
    const int*   agent_id = (const int*)d_in[17];
    float* out = (float*)d_out;

    char* ws = (char*)d_ws;
    __bf16* WmT   = (__bf16*)(ws + OFF_WMT);
    __bf16* Wo1T  = (__bf16*)(ws + OFF_WO1T);
    __bf16* Pcat  = (__bf16*)(ws + OFF_PCAT);
    float*  selfo = (float*) (ws + OFF_SELF);
    __bf16* Xb    = (__bf16*)(ws + OFF_X);
    __bf16* Yb    = (__bf16*)(ws + OFF_Y);

    fold_wmt<<<KC, CH, 0, stream>>>(W_ha2, b_ha2, W_hv2, b_hv2, merger_w, WmT);
    transpose_o1<<<CH, CH, 0, stream>>>(W_o1, Wo1T);
    feat_kernel<<<BATCH, 128, 0, stream>>>(state, action, W_self, b_self,
                                           W_ha1, b_ha1, W_hv1, b_hv1,
                                           agent_id, Pcat, selfo);
    wmma_gemm<<<dim3(BATCH / 16, CH / 64), 128, 0, stream>>>(Pcat, WmT, KC, selfo, nullptr, Xb);
    wmma_gemm<<<dim3(BATCH / 16, CH / 64), 128, 0, stream>>>(Xb, Wo1T, CH, nullptr, b_o1, Yb);
    out_gemv<<<BATCH / 4, 128, 0, stream>>>(Yb, W_o2, b_o2, out);
}
